// SAModule_10617159156371
// MI455X (gfx1250) — matrix operand order repaired and checked by reading, not compiled, not run on hardware
//
#include <hip/hip_runtime.h>

typedef __attribute__((ext_vector_type(16))) _Float16 v16h;
typedef __attribute__((ext_vector_type(8)))  float    v8f;

#define B_     16
#define NPC_   4096
#define MPC_   1024
#define MTOT_  (B_ * MPC_)   // 16384 centroids
#define KNN_   64
#define FIN_   64
#define KPAD_  96            // 67 features padded to 3 K-tiles of 32
#define H1_    64
#define H2_    128
#define R2_    0.04f         // R = 0.2

// ---------------------------------------------------------------------------
// Kernel 1: farthest point sampling, one block per cloud.
// Emits idx[m] = last BEFORE the argmax update (matches the lax.scan carry).
// ---------------------------------------------------------------------------
__global__ void __launch_bounds__(256) fps_kernel(const float* __restrict__ pos,
                                                  int* __restrict__ idx_out,
                                                  float* __restrict__ pos_c,
                                                  int* __restrict__ batch_out) {
    const int cloud = blockIdx.x;
    const int tid   = threadIdx.x;
    const int base  = cloud * NPC_;

    __shared__ float d[NPC_];
    __shared__ float rv[256];
    __shared__ int   ri[256];
    __shared__ int   s_last;
    __shared__ float s_lx, s_ly, s_lz;

    for (int i = tid; i < NPC_; i += 256) d[i] = 3.0e38f;
    for (int m = tid; m < MPC_; m += 256) batch_out[cloud * MPC_ + m] = cloud;
    if (tid == 0) s_last = 0;
    __syncthreads();

    for (int m = 0; m < MPC_; ++m) {
        if (tid == 0) {
            const int last = s_last;
            const float lx = pos[(base + last) * 3 + 0];
            const float ly = pos[(base + last) * 3 + 1];
            const float lz = pos[(base + last) * 3 + 2];
            const int c = cloud * MPC_ + m;
            idx_out[c] = base + last;          // absolute point index
            pos_c[c * 3 + 0] = lx;
            pos_c[c * 3 + 1] = ly;
            pos_c[c * 3 + 2] = lz;
            s_lx = lx; s_ly = ly; s_lz = lz;
        }
        __syncthreads();
        const float lx = s_lx, ly = s_ly, lz = s_lz;

        float bv = -1.0f; int bi = 0;
        for (int i = tid; i < NPC_; i += 256) {
            const float dx = pos[(base + i) * 3 + 0] - lx;
            const float dy = pos[(base + i) * 3 + 1] - ly;
            const float dz = pos[(base + i) * 3 + 2] - lz;
            const float nd = fminf(d[i], dx * dx + dy * dy + dz * dz);
            d[i] = nd;
            if (nd > bv) { bv = nd; bi = i; }  // strided visit: keeps lowest index on ties
        }
        rv[tid] = bv; ri[tid] = bi;
        __syncthreads();
        for (int s = 128; s > 0; s >>= 1) {
            if (tid < s) {
                const float ov = rv[tid + s]; const int oi = ri[tid + s];
                if (ov > rv[tid] || (ov == rv[tid] && oi < ri[tid])) { rv[tid] = ov; ri[tid] = oi; }
            }
            __syncthreads();
        }
        if (tid == 0) s_last = ri[0];
        __syncthreads();
    }
}

// ---------------------------------------------------------------------------
// Kernel 2: radius top-K via histogram threshold. One block per centroid.
// Valid neighbors are compacted into slots [0, cnt); rest filled with self.
// ---------------------------------------------------------------------------
__global__ void __launch_bounds__(256) knn_kernel(const float* __restrict__ pos,
                                                  const float* __restrict__ pos_c,
                                                  const int* __restrict__ idx,
                                                  int* __restrict__ nidx,
                                                  int* __restrict__ cnt_out) {
    const int c    = blockIdx.x;
    const int tid  = threadIdx.x;
    const int base = (c >> 10) * NPC_;
    const float cx = pos_c[c * 3 + 0];
    const float cy = pos_c[c * 3 + 1];
    const float cz = pos_c[c * 3 + 2];

    __shared__ int hist[128];
    __shared__ int s_cut, s_below, s_lo, s_eq;

    for (int i = tid; i < 128; i += 256) hist[i] = 0;
    if (tid == 0) { s_lo = 0; s_eq = 0; }
    __syncthreads();

    const float scale = 128.0f / R2_;
    for (int i = tid; i < NPC_; i += 256) {
        const float dx = pos[(base + i) * 3 + 0] - cx;
        const float dy = pos[(base + i) * 3 + 1] - cy;
        const float dz = pos[(base + i) * 3 + 2] - cz;
        const float d2 = dx * dx + dy * dy + dz * dz;
        if (d2 <= R2_) {
            int b = (int)(d2 * scale); if (b > 127) b = 127;
            atomicAdd(&hist[b], 1);
        }
    }
    __syncthreads();
    if (tid == 0) {
        int cum = 0, cut = 128;
        for (int b = 0; b < 128; ++b) {
            if (cum + hist[b] >= KNN_) { cut = b; break; }
            cum += hist[b];
        }
        s_cut = cut; s_below = cum;
    }
    __syncthreads();
    const int cut = s_cut, below = s_below;

    for (int i = tid; i < NPC_; i += 256) {
        const float dx = pos[(base + i) * 3 + 0] - cx;
        const float dy = pos[(base + i) * 3 + 1] - cy;
        const float dz = pos[(base + i) * 3 + 2] - cz;
        const float d2 = dx * dx + dy * dy + dz * dz;
        if (d2 <= R2_) {
            int b = (int)(d2 * scale); if (b > 127) b = 127;
            if (b < cut) {
                const int s = atomicAdd(&s_lo, 1);
                nidx[c * KNN_ + s] = base + i;
            } else if (b == cut) {
                const int s = atomicAdd(&s_eq, 1);
                const int slot = below + s;
                if (slot < KNN_) nidx[c * KNN_ + slot] = base + i;
            }
        }
    }
    __syncthreads();
    int tot = below + s_eq; if (tot > KNN_) tot = KNN_;
    if (tid == 0) cnt_out[c] = tot;
    const int self = idx[c];
    for (int s = tot + tid; s < KNN_; s += 256) nidx[c * KNN_ + s] = self;
}

// ---------------------------------------------------------------------------
// Kernel 3: weight precondition — fp32 -> f16, TRANSPOSED [N][K] so B-fragment
// loads become two contiguous 16-byte runs per lane. W1 K-padded 67 -> 96.
// ---------------------------------------------------------------------------
__global__ void __launch_bounds__(256) wconv_kernel(const float* __restrict__ W1,
                                                    const float* __restrict__ W2,
                                                    _Float16* __restrict__ W1t,
                                                    _Float16* __restrict__ W2t) {
    const int t = blockIdx.x * 256 + threadIdx.x;
    if (t < H1_ * KPAD_) {           // W1t[n][k], n<64, k<96
        const int n = t / KPAD_, k = t - n * KPAD_;
        W1t[t] = (k < FIN_ + 3) ? (_Float16)W1[k * H1_ + n] : (_Float16)0.0f;
    }
    if (t < H2_ * H1_) {             // W2t[n][k], n<128, k<64
        const int n = t / H1_, k = t - n * H1_;
        W2t[t] = (_Float16)W2[k * H2_ + n];
    }
}

// ---------------------------------------------------------------------------
// WMMA fragment loaders (ISA 7.12.2 16-bit layouts), vectorized:
// A 16x32: lane row = l15; halves = two contiguous 8-half runs
//          (k = kbase+8*hi+0..7 and k = kbase+16+8*hi+0..7) -> 2x b128 loads.
// B 32x16 from TRANSPOSED [N][K] store: lane column n = nt*16+l15;
//          halves k = kbase+8*hi+0..7 and k = kbase+16+8*hi+0..7 -> 2x b128.
// ---------------------------------------------------------------------------
__device__ inline v16h load_a_frag(const _Float16* base, int ld, int ktile,
                                   int l15, int hi) {
    const _Float16* p = base + l15 * ld + ktile * 32 + hi * 8;
    union { v16h v; float4 q[2]; } u;
    u.q[0] = *(const float4*)(p);
    u.q[1] = *(const float4*)(p + 16);
    return u.v;
}

__device__ inline v16h load_bt_frag(const _Float16* baseT, int ldk, int ktile,
                                    int ntile, int l15, int hi) {
    const _Float16* p = baseT + (ntile * 16 + l15) * ldk + ktile * 32 + hi * 8;
    union { v16h v; float4 q[2]; } u;
    u.q[0] = *(const float4*)(p);
    u.q[1] = *(const float4*)(p + 16);
    return u.v;
}

// ---------------------------------------------------------------------------
// Kernel 4: fused gather -> MLP(67->64 relu ->128) -> masked max.
// One block (4 waves) per centroid; each wave owns a 16-edge row tile.
// Biases folded into the WMMA C initializer (lane = N column in C layout).
// ---------------------------------------------------------------------------
__global__ void __launch_bounds__(128) mlp_kernel(
    const float* __restrict__ x, const float* __restrict__ pos,
    const float* __restrict__ pos_c, const int* __restrict__ nidx,
    const int* __restrict__ cnt, const _Float16* __restrict__ W1t,
    const float* __restrict__ b1, const _Float16* __restrict__ W2t,
    const float* __restrict__ b2, float* __restrict__ out) {
    __shared__ __align__(16) _Float16 sW1[H1_ * KPAD_];   // 12 KB, [n][k]
    __shared__ __align__(16) _Float16 sW2[H2_ * H1_];     // 16 KB, [n][k]
    __shared__ __align__(16) _Float16 sA[4][16 * KPAD_];  // 12 KB, [row][k]
    __shared__ __align__(16) _Float16 sH1[4][16 * H1_];   //  8 KB, [row][k]
    __shared__ float sPart[4][H2_];                       //  2 KB

    const int c    = blockIdx.x;
    const int tid  = threadIdx.x;
    const int w    = tid >> 5;
    const int lane = tid & 31;
    const int l15  = lane & 15;
    const int hi   = lane >> 4;

    __builtin_prefetch(W1t, 0, 0);  // global_prefetch_b8 into near caches
    __builtin_prefetch(W2t, 0, 0);

    // stage weights via 16-byte copies: global_load_b128 + ds_store_b128
    {
        const uint4* g1 = (const uint4*)W1t; uint4* l1 = (uint4*)sW1;
        for (int t = tid; t < (H1_ * KPAD_) / 8; t += 128) l1[t] = g1[t];
        const uint4* g2 = (const uint4*)W2t; uint4* l2 = (uint4*)sW2;
        for (int t = tid; t < (H2_ * H1_) / 8; t += 128) l2[t] = g2[t];
    }

    const float px = pos_c[c * 3 + 0];
    const float py = pos_c[c * 3 + 1];
    const float pz = pos_c[c * 3 + 2];

    // gather edge features [x_j, pos_j - pos_i, 0-pad] into this wave's A tile,
    // packed 4 halves at a time (float4 global read, b64 LDS store).
    for (int t = lane; t < 16 * (KPAD_ / 4); t += 32) {
        const int row = t / (KPAD_ / 4), g = t - row * (KPAD_ / 4);
        const int j = nidx[c * KNN_ + w * 16 + row];
        union { _Float16 h[4]; uint2 u; } pk;
        if (g < FIN_ / 4) {
            const float4 v = *(const float4*)(x + (size_t)j * FIN_ + g * 4);
            pk.h[0] = (_Float16)v.x; pk.h[1] = (_Float16)v.y;
            pk.h[2] = (_Float16)v.z; pk.h[3] = (_Float16)v.w;
        } else if (g == FIN_ / 4) {
            pk.h[0] = (_Float16)(pos[j * 3 + 0] - px);
            pk.h[1] = (_Float16)(pos[j * 3 + 1] - py);
            pk.h[2] = (_Float16)(pos[j * 3 + 2] - pz);
            pk.h[3] = (_Float16)0.0f;
        } else {
            pk.u = make_uint2(0u, 0u);
        }
        *(uint2*)(&sA[w][row * KPAD_ + g * 4]) = pk.u;
    }
    __syncthreads();

    // ---- layer 1: [16 x 96] @ [96 x 64] + b1, relu ----
    v8f acc1[4];
#pragma unroll
    for (int nt = 0; nt < 4; ++nt) {
        const float bv = b1[nt * 16 + l15];
#pragma unroll
        for (int r = 0; r < 8; ++r) acc1[nt][r] = bv;
    }
#pragma unroll
    for (int kt = 0; kt < 3; ++kt) {
        const v16h a = load_a_frag(&sA[w][0], KPAD_, kt, l15, hi);
#pragma unroll
        for (int nt = 0; nt < 4; ++nt) {
            const v16h b = load_bt_frag(sW1, KPAD_, kt, nt, l15, hi);
            acc1[nt] = __builtin_amdgcn_wmma_f32_16x16x32_f16(
                false, a, false, b, (short)0, acc1[nt], false, false);
        }
    }
    // relu + f16 restage with D layout: (r, lane) -> M = r + hi*8, N = nt*16 + l15
#pragma unroll
    for (int nt = 0; nt < 4; ++nt)
#pragma unroll
        for (int r = 0; r < 8; ++r) {
            float v = acc1[nt][r]; v = v > 0.0f ? v : 0.0f;
            sH1[w][(r + hi * 8) * H1_ + nt * 16 + l15] = (_Float16)v;
        }
    __syncthreads();

    // ---- layer 2: [16 x 64] @ [64 x 128] + b2 ----
    v8f acc2[8];
#pragma unroll
    for (int nt = 0; nt < 8; ++nt) {
        const float bv = b2[nt * 16 + l15];
#pragma unroll
        for (int r = 0; r < 8; ++r) acc2[nt][r] = bv;
    }
#pragma unroll
    for (int kt = 0; kt < 2; ++kt) {
        const v16h a = load_a_frag(&sH1[w][0], H1_, kt, l15, hi);
#pragma unroll
        for (int nt = 0; nt < 8; ++nt) {
            const v16h b = load_bt_frag(sW2, H1_, kt, nt, l15, hi);
            acc2[nt] = __builtin_amdgcn_wmma_f32_16x16x32_f16(
                false, a, false, b, (short)0, acc2[nt], false, false);
        }
    }

    // ---- masked max over the wave's 16 edges, then over 4 waves ----
    const int vc = cnt[c];
#pragma unroll
    for (int nt = 0; nt < 8; ++nt) {
        float m = -3.0e38f;
#pragma unroll
        for (int r = 0; r < 8; ++r) {
            const int edge = w * 16 + r + hi * 8;
            const float v = (edge < vc) ? acc2[nt][r] : -3.0e38f;
            m = fmaxf(m, v);
        }
        m = fmaxf(m, __shfl_xor(m, 16, 32));  // fold other half (other 8 rows, same N)
        if (hi == 0) sPart[w][nt * 16 + l15] = m;
    }
    __syncthreads();
    if (tid < H2_) {
        const float m = fmaxf(fmaxf(sPart[0][tid], sPart[1][tid]),
                              fmaxf(sPart[2][tid], sPart[3][tid]));
        out[c * H2_ + tid] = (vc > 0) ? m : 0.0f;
    }
}

// ---------------------------------------------------------------------------
// Launcher. Inputs: x, pos, batch, W1, b1, W2, b2.
// d_out = [out 16384x128 f32][pos_out 16384x3 f32][batch_out 16384 i32].
// ---------------------------------------------------------------------------
extern "C" void kernel_launch(void* const* d_in, const int* in_sizes, int n_in,
                              void* d_out, int out_size, void* d_ws, size_t ws_size,
                              hipStream_t stream) {
    const float* x   = (const float*)d_in[0];
    const float* pos = (const float*)d_in[1];
    const float* W1  = (const float*)d_in[3];
    const float* b1  = (const float*)d_in[4];
    const float* W2  = (const float*)d_in[5];
    const float* b2  = (const float*)d_in[6];

    float* out       = (float*)d_out;                     // [16384,128]
    float* pos_out   = out + (size_t)MTOT_ * H2_;         // [16384,3]
    int*   batch_out = (int*)(pos_out + (size_t)MTOT_ * 3);

    char* ws = (char*)d_ws;
    int* idx  = (int*)ws;            ws += (size_t)MTOT_ * sizeof(int);
    int* nidx = (int*)ws;            ws += (size_t)MTOT_ * KNN_ * sizeof(int);
    int* cnt  = (int*)ws;            ws += (size_t)MTOT_ * sizeof(int);
    _Float16* W1t = (_Float16*)ws;   ws += (size_t)H1_ * KPAD_ * sizeof(_Float16);
    _Float16* W2t = (_Float16*)ws;

    fps_kernel<<<B_, 256, 0, stream>>>(pos, idx, pos_out, batch_out);
    wconv_kernel<<<(H2_ * H1_ + 255) / 256, 256, 0, stream>>>(W1, W2, W1t, W2t);
    knn_kernel<<<MTOT_, 256, 0, stream>>>(pos, pos_out, idx, nidx, cnt);
    mlp_kernel<<<MTOT_, 128, 0, stream>>>(x, pos, pos_out, nidx, cnt,
                                          W1t, b1, W2t, b2, out);
}
